// MultiHeadAttentionLayer_49495203119600
// MI455X (gfx1250) — matrix operand order repaired and checked by reading, not compiled, not run on hardware
//
#include <hip/hip_runtime.h>
#include <hip/hip_bf16.h>

// ---------------------------------------------------------------------------
// T5 multi-head self-attention for MI455X (gfx1250), bf16 WMMA + flash softmax
// B=4, S=2048, D_MODEL=512, H=8, D_KV=64, NUM_BUCKETS=32, MAX_DISTANCE=128
// Attention computed transposed (S^T = K.Q^T, O^T = V^T.P^T) so the softmax
// dimension is per-lane: scalar running stats, packed LDS stores, vector
// epilogue stores.  K/V tiles staged with double-buffered
// global_load_async_to_lds_b128 (ASYNCcnt).
// ---------------------------------------------------------------------------

typedef __attribute__((ext_vector_type(16))) __bf16 vbf16;   // 8 VGPRs
typedef __attribute__((ext_vector_type(8)))  __bf16 vbf8;    // 4 VGPRs
typedef __attribute__((ext_vector_type(8)))  float  vf8;     // 8 VGPRs
typedef __attribute__((ext_vector_type(4)))  float  vf4;     // 4 VGPRs

#define BB 4
#define SS 2048
#define DM 512
#define HH 8
#define DK 64
#define LOG2E 1.44269504088896f

// Async copy of 32 contiguous bytes global->LDS (2x b128; the 24-bit inst
// offset is added to BOTH the global and LDS addresses per ISA 08 §4.4).
#define ASYNC_LDS_32B(ldsptr, gptr)                                          \
    asm volatile("global_load_async_to_lds_b128 %0, %1, off\n\t"             \
                 "global_load_async_to_lds_b128 %0, %1, off offset:16"       \
                 :: "v"((unsigned int)(uintptr_t)(ldsptr)),                  \
                    "v"((unsigned long long)(uintptr_t)(gptr))               \
                 : "memory")

// ---------------------------------------------------------------------------
// Kernel 1: fused QKV projection  P = X @ W  (bf16 out, f32 accumulate)
// grid = (M/128, H, 3) ; block = 256 (8 waves). z picks {q,k,v}.
// q,k stored [B,H,S,D]; v stored transposed [B,H,D,S] (A-operand of V^T.P^T).
// ---------------------------------------------------------------------------
__global__ __launch_bounds__(256)
void qkv_proj_kernel(const float* __restrict__ q_in, const float* __restrict__ k_in,
                     const float* __restrict__ v_in,
                     const float* __restrict__ Wq, const float* __restrict__ Wk,
                     const float* __restrict__ Wv,
                     __bf16* __restrict__ qbf, __bf16* __restrict__ kbf,
                     __bf16* __restrict__ vbf)
{
    __shared__ __bf16 alds[128][32];   // X tile (M x K), bf16
    __shared__ __bf16 wlds[64][32];    // W tile transposed (N x K), bf16

    const int z = blockIdx.z;
    const float* X  = (z == 0) ? q_in : (z == 1) ? k_in : v_in;
    const float* W  = (z == 0) ? Wq   : (z == 1) ? Wk   : Wv;
    __bf16*      O  = (z == 0) ? qbf  : (z == 1) ? kbf  : vbf;

    const int t     = threadIdx.x;
    const int lane  = t & 31;
    const int wave  = t >> 5;
    const int half  = lane >> 4;
    const int l16   = lane & 15;
    const int mbase = blockIdx.x * 128;
    const int h     = blockIdx.y;          // head -> N tile
    const int nbase = h * DK;

    vf8 acc[4];
#pragma unroll
    for (int j = 0; j < 4; ++j)
#pragma unroll
        for (int r = 0; r < 8; ++r) acc[j][r] = 0.0f;

    for (int ks = 0; ks < DM; ks += 32) {
        __syncthreads();
        {   // X tile: 128 x 32, f32 -> bf16, 64B per thread
            const int row = t >> 1;
            const int c0  = (t & 1) * 16;
            const float* src = X + (size_t)(mbase + row) * DM + ks + c0;
#pragma unroll
            for (int i = 0; i < 16; ++i) alds[row][c0 + i] = (__bf16)src[i];
            if (ks + 32 < DM)
                __builtin_prefetch(src + 32, 0, 1);   // global_prefetch next K-step
        }
        {   // W tile: 32(K) x 64(N) -> transposed LDS [n][k]
            const int kr = t >> 3;
            const int c0 = (t & 7) * 8;
            const float* src = W + (size_t)(ks + kr) * DM + nbase + c0;
#pragma unroll
            for (int i = 0; i < 8; ++i) wlds[c0 + i][kr] = (__bf16)src[i];
        }
        __syncthreads();

        // A fragment: lane row = 16*wave + l16 ; e <-> k = 16*half + e
        vbf16 a;
        {
            vbf8 lo = *(const vbf8*)&alds[16 * wave + l16][16 * half];
            vbf8 hi = *(const vbf8*)&alds[16 * wave + l16][16 * half + 8];
#pragma unroll
            for (int i = 0; i < 8; ++i) { a[i] = lo[i]; a[i + 8] = hi[i]; }
        }
#pragma unroll
        for (int j = 0; j < 4; ++j) {
            // B fragment: lane n = 16*j + l16 ; e <-> k = 16*half + e (32B contiguous)
            vbf16 bm = *(const vbf16*)&wlds[16 * j + l16][16 * half];
            acc[j] = __builtin_amdgcn_wmma_f32_16x16x32_bf16(
                         false, a, false, bm, (short)0, acc[j], false, false);
        }
    }

    // store: C frag element r -> row (r + 8*half), col (16*j + l16)
    const int b = mbase >> 11;                     // 2048 rows per batch, tile-aligned
#pragma unroll
    for (int j = 0; j < 4; ++j)
#pragma unroll
        for (int r = 0; r < 8; ++r) {
            const int m = mbase + 16 * wave + r + 8 * half;
            const int s = m & (SS - 1);
            const int d = 16 * j + l16;
            const __bf16 vv = (__bf16)acc[j][r];
            if (z < 2) O[(((size_t)b * HH + h) * SS + s) * DK + d] = vv; // [B,H,S,D]
            else       O[(((size_t)b * HH + h) * DK + d) * SS + s] = vv; // [B,H,D,S]
        }
}

// ---------------------------------------------------------------------------
// Kernel 2: relative-position bias table  bias_tab[h][delta+2047], delta in
// [-2047,2047].  T5 bucketing, bidirectional, NUM_BUCKETS=32, MAX_DISTANCE=128:
// val_if_large = 8 + 2*log2(|d|/8)  (log(x/8)/log(16)*8 == 2*log2(x/8))
// ---------------------------------------------------------------------------
__global__ void bias_table_kernel(const float* __restrict__ rel_emb,
                                  float* __restrict__ bias_tab)
{
    const int idx = blockIdx.x * blockDim.x + threadIdx.x;
    if (idx >= 4096) return;
    const int delta = idx - 2047;
    const int ret   = (delta > 0) ? 16 : 0;
    const int a     = (delta < 0) ? -delta : delta;
    int bucket;
    if (a < 8) {
        bucket = ret + a;
    } else {
        const float l2 = __builtin_amdgcn_logf((float)a * 0.125f);   // v_log_f32
        int v = 8 + (int)(l2 * 2.0f);
        if (v > 15) v = 15;
        bucket = ret + v;
    }
#pragma unroll
    for (int h = 0; h < HH; ++h)
        bias_tab[h * 4096 + idx] = rel_emb[bucket * HH + h];
}

// ---------------------------------------------------------------------------
// Kernel 3: transposed flash attention.  grid = (S/128, B*H); block = 256.
// Each wave owns 16 q-columns.  S^T frags: rows = keys, cols = q (per-lane).
// Double-buffered async K/V staging; 16 WMMAs per 64-key tile per wave.
// ---------------------------------------------------------------------------
__global__ __launch_bounds__(256)
void t5_flash_attn_kernel(const __bf16* __restrict__ qbf, const __bf16* __restrict__ kbf,
                          const __bf16* __restrict__ vbf, const float* __restrict__ key_mask,
                          const float* __restrict__ bias_tab, float* __restrict__ out)
{
    __shared__ __bf16 ktile[2][64][64];   // [buf][key][d]   16 KB
    __shared__ __bf16 vtile[2][64][64];   // [buf][d][key]   16 KB (v pre-transposed)
    __shared__ __bf16 ptile[8][16][64];   // [wave][qcol][key] P^T scratch, 16 KB

    const int t    = threadIdx.x;
    const int lane = t & 31;
    const int wave = t >> 5;
    const int half = lane >> 4;
    const int l16  = lane & 15;
    const int bh   = blockIdx.y;
    const int b    = bh >> 3;
    const int h    = bh & 7;
    const int qrw  = blockIdx.x * 128 + 16 * wave;   // this wave's first q row
    const int qp   = qrw + l16;                      // this lane's q position

    const __bf16* kbase = kbf + (size_t)bh * SS * DK;   // [key][d]
    const __bf16* vbase = vbf + (size_t)bh * DK * SS;   // [d][key]
    const float*  mbase = key_mask + (size_t)b * SS;
    const float*  btab  = bias_tab + (size_t)h * 4096;

    // Q^T B-fragments: lane n = q column, elements contiguous in d (32B loads)
    vbf16 qbm[2];
    {
        const __bf16* qrow = qbf + ((size_t)bh * SS + qp) * DK;
#pragma unroll
        for (int kk = 0; kk < 2; ++kk)
            qbm[kk] = *(const vbf16*)(qrow + 32 * kk + 16 * half);
    }

    float m_run = -1e30f, l_run = 0.0f;        // per-lane (per-q-row) stats
    vf8 ot[4];                                  // O^T: rows d = 16*jj + r + 8*half
#pragma unroll
    for (int j = 0; j < 4; ++j)
#pragma unroll
        for (int r = 0; r < 8; ++r) ot[j][r] = 0.0f;

    // async stage of one 64-key tile: 4 async instructions per wave
    auto stage_async = [&](int kb, int buf) {
        const int row = t >> 2;
        const int c0  = (t & 3) * 16;
        ASYNC_LDS_32B(&ktile[buf][row][c0], kbase + (size_t)(kb + row) * DK + c0);
        ASYNC_LDS_32B(&vtile[buf][row][c0], vbase + (size_t)row * SS + kb + c0);
    };

    stage_async(0, 0);                          // prologue

    for (int kb = 0; kb < SS; kb += 64) {
        const int ib = (kb >> 6) & 1;
        if (kb + 64 < SS) {
            stage_async(kb + 64, ib ^ 1);       // overlap next tile with compute
            asm volatile("s_wait_asynccnt 4" ::: "memory");  // oldest 4 done
        } else {
            asm volatile("s_wait_asynccnt 0" ::: "memory");
        }
        __syncthreads();                        // tile ib resident for all waves

        // ----- S^T = K . Q^T : 4 frags (16 keys x 16 q), K-dim = DK -----
        vf8 st[4];
#pragma unroll
        for (int jj = 0; jj < 4; ++jj) {
            vf8 c;
#pragma unroll
            for (int r = 0; r < 8; ++r) c[r] = 0.0f;
#pragma unroll
            for (int kk = 0; kk < 2; ++kk) {
                vbf16 ka;   // A frag: lane row = key 16*jj + l16, elems contiguous d
                vbf8 lo = *(const vbf8*)&ktile[ib][16 * jj + l16][32 * kk + 16 * half];
                vbf8 hi = *(const vbf8*)&ktile[ib][16 * jj + l16][32 * kk + 16 * half + 8];
#pragma unroll
                for (int i = 0; i < 8; ++i) { ka[i] = lo[i]; ka[i + 8] = hi[i]; }
                c = __builtin_amdgcn_wmma_f32_16x16x32_bf16(
                        false, ka, false, qbm[kk], (short)0, c, false, false);
            }
            st[jj] = c;
        }

        // ----- bias + additive key mask (element r <-> key kb+16*jj+r+8*half) -----
#pragma unroll
        for (int jj = 0; jj < 4; ++jj)
#pragma unroll
            for (int r = 0; r < 8; ++r) {
                const int key = kb + 16 * jj + r + 8 * half;
                st[jj][r] += btab[key - qp + 2047]
                           + (1.0f - mbase[key]) * -10000.0f;
            }

        // ----- online softmax, per-lane scalar stats -----
        float tmax = -1e30f;
#pragma unroll
        for (int jj = 0; jj < 4; ++jj)
#pragma unroll
            for (int r = 0; r < 8; ++r) tmax = fmaxf(tmax, st[jj][r]);
        tmax = fmaxf(tmax, __shfl_xor(tmax, 16, 32));   // merge the two halves

        const float mnew  = fmaxf(m_run, tmax);
        const float scale = __builtin_amdgcn_exp2f((m_run - mnew) * LOG2E);
        m_run = mnew;

        float rsum = 0.0f;
#pragma unroll
        for (int jj = 0; jj < 4; ++jj)
#pragma unroll
            for (int r = 0; r < 8; ++r) {
                const float p = __builtin_amdgcn_exp2f((st[jj][r] - mnew) * LOG2E);
                st[jj][r] = p;
                rsum += p;
            }
        rsum += __shfl_xor(rsum, 16, 32);
        l_run = l_run * scale + rsum;

#pragma unroll
        for (int jj = 0; jj < 4; ++jj)
#pragma unroll
            for (int r = 0; r < 8; ++r) ot[jj][r] *= scale;

        // ----- P^T -> LDS: 8 consecutive keys per lane = one packed 16B store -----
#pragma unroll
        for (int jj = 0; jj < 4; ++jj) {
            vbf8 pk;
#pragma unroll
            for (int r = 0; r < 8; ++r) pk[r] = (__bf16)st[jj][r];
            *(vbf8*)&ptile[wave][l16][16 * jj + 8 * half] = pk;
        }
        // LDS is in-order within a wave: store->load needs no extra sync

        // ----- O^T += V^T . P^T  (K-dim = 64 keys = 2 chunks) -----
#pragma unroll
        for (int c = 0; c < 2; ++c) {
            vbf16 pb = *(const vbf16*)&ptile[wave][l16][32 * c + 16 * half];
#pragma unroll
            for (int jj = 0; jj < 4; ++jj) {
                vbf16 va;   // A frag: lane row = d 16*jj + l16, elems contiguous keys
                vbf8 lo = *(const vbf8*)&vtile[ib][16 * jj + l16][32 * c + 16 * half];
                vbf8 hi = *(const vbf8*)&vtile[ib][16 * jj + l16][32 * c + 16 * half + 8];
#pragma unroll
                for (int i = 0; i < 8; ++i) { va[i] = lo[i]; va[i + 8] = hi[i]; }
                ot[jj] = __builtin_amdgcn_wmma_f32_16x16x32_bf16(
                             false, va, false, pb, (short)0, ot[jj], false, false);
            }
        }
        __syncthreads();    // all waves done with tile ib before it is reissued
    }

    // ----- epilogue: per-lane normalize; d contiguous per lane -> 16B stores -----
    const float inv = 1.0f / l_run;
    float* obase = out + ((size_t)b * SS + qp) * (HH * DK) + h * DK + 8 * half;
#pragma unroll
    for (int jj = 0; jj < 4; ++jj) {
        vf4 o0, o1;
#pragma unroll
        for (int r = 0; r < 4; ++r) { o0[r] = ot[jj][r] * inv; o1[r] = ot[jj][r + 4] * inv; }
        *(vf4*)(obase + 16 * jj)     = o0;
        *(vf4*)(obase + 16 * jj + 4) = o1;
    }
}

// ---------------------------------------------------------------------------
// Launch
// ---------------------------------------------------------------------------
extern "C" void kernel_launch(void* const* d_in, const int* in_sizes, int n_in,
                              void* d_out, int out_size, void* d_ws, size_t ws_size,
                              hipStream_t stream)
{
    const float* query    = (const float*)d_in[0];
    const float* key      = (const float*)d_in[1];
    const float* value    = (const float*)d_in[2];
    const float* key_mask = (const float*)d_in[3];
    const float* Wq       = (const float*)d_in[4];
    const float* Wk       = (const float*)d_in[5];
    const float* Wv       = (const float*)d_in[6];
    const float* rel_emb  = (const float*)d_in[7];
    // d_in[8] = start_pos (always 0 in this configuration)

    char* ws = (char*)d_ws;
    const size_t QKV_BYTES = (size_t)BB * HH * SS * DK * sizeof(__bf16); // 8 MB each
    __bf16* qbf      = (__bf16*)(ws);
    __bf16* kbf      = (__bf16*)(ws + QKV_BYTES);
    __bf16* vbf      = (__bf16*)(ws + 2 * QKV_BYTES);
    float*  bias_tab = (float*) (ws + 3 * QKV_BYTES);                    // 128 KB
    float*  out      = (float*)d_out;

    bias_table_kernel<<<16, 256, 0, stream>>>(rel_emb, bias_tab);
    qkv_proj_kernel<<<dim3(64, HH, 3), 256, 0, stream>>>(
        query, key, value, Wq, Wk, Wv, qbf, kbf, vbf);
    t5_flash_attn_kernel<<<dim3(SS / 128, BB * HH), 256, 0, stream>>>(
        qbf, kbf, vbf, key_mask, bias_tab, out);
}
